// PersuasionGNN_83580063580856
// MI455X (gfx1250) — matrix operand order repaired and checked by reading, not compiled, not run on hardware
//
#include <hip/hip_runtime.h>
#include <hip/hip_bf16.h>

typedef __bf16 bf16;
typedef __attribute__((ext_vector_type(8)))  __bf16 v8bf;
typedef __attribute__((ext_vector_type(16))) __bf16 v16bf;
typedef __attribute__((ext_vector_type(8)))  float  v8f;

#define NN  10000
#define EE  160000
#define DIN 768
#define HID 256
#define DD3 128
#define NH  4
#define DG  512   // NH * DD3

static inline int cdiv_h(long long a, long long b) { return (int)((a + b - 1) / b); }

// ---------- ordered-float encoding for atomic max ----------
__device__ __forceinline__ unsigned f2ord(float x) {
  unsigned u = __float_as_uint(x);
  return (u & 0x80000000u) ? ~u : (u | 0x80000000u);
}
__device__ __forceinline__ float ord2f(unsigned k) {
  unsigned u = (k & 0x80000000u) ? (k & 0x7fffffffu) : ~k;
  return __uint_as_float(u);
}
__device__ __forceinline__ float lrelu(float v) { return v > 0.f ? v : 0.2f * v; }

union FragU { v16bf v; v8bf h[2]; };

// ---------- WMMA GEMM: C = act(A@B^T + bias), 16x64 C strip per wave ----------
// A: [M][K] bf16 row-major.  BT: [Nout][K] bf16 (i.e. W^T).  K must be a multiple of 64.
// Addressing: uniform SGPR base + per-lane 32-bit byte offsets (SADDR-form global loads).
__global__ void gemm_bf16_wmma(const bf16* __restrict__ A, const bf16* __restrict__ BT,
                               const float* __restrict__ bias, int relu,
                               float* __restrict__ Cf, int M, int Nout, int K) {
  int ngrp = Nout >> 6;                                      // groups of 4 N-tiles
  int tile = blockIdx.x * (blockDim.x >> 5) + (threadIdx.x >> 5);
  if (tile >= (M >> 4) * ngrp) return;                       // whole wave exits together
  int tm = tile / ngrp, tg = tile - tm * ngrp;
  int lane = threadIdx.x & 31;
  int half = lane >> 4, l15 = lane & 15;
  int m0 = tm << 4, n0 = tg << 6;

  v8f acc[4];
  #pragma unroll
  for (int j = 0; j < 4; ++j) acc[j] = (v8f){0.f,0.f,0.f,0.f,0.f,0.f,0.f,0.f};

  const char* Ab = (const char*)A;
  const char* Bb = (const char*)BT;
  unsigned aofs = (unsigned)(((m0 + l15) * K + half * 8) * 2);
  unsigned bofs[4];
  #pragma unroll
  for (int j = 0; j < 4; ++j)
    bofs[j] = (unsigned)(((n0 + 16 * j + l15) * K + half * 8) * 2);

  auto loadA = [&](FragU& a, int k0) {
    unsigned o = aofs + (unsigned)k0 * 2;
    a.h[0] = *(const v8bf*)(Ab + o);           // K = k0 + half*8 + 0..7
    a.h[1] = *(const v8bf*)(Ab + o + 32);      // K = k0 + 16 + half*8 + 0..7
  };
  auto loadB = [&](FragU* b, int k0) {
    #pragma unroll
    for (int j = 0; j < 4; ++j) {
      unsigned o = bofs[j] + (unsigned)k0 * 2;
      b[j].h[0] = *(const v8bf*)(Bb + o);      // K = k0 + half*8 + 0..7
      b[j].h[1] = *(const v8bf*)(Bb + o + 32); // K = k0 + 16 + half*8 + 0..7
    }
  };

  FragU a0, a1, b0[4], b1[4];
  loadA(a0, 0);
  loadB(b0, 0);
  for (int k0 = 0; k0 < K; k0 += 64) {           // K % 64 == 0
    loadA(a1, k0 + 32);                          // prefetch stage 1
    loadB(b1, k0 + 32);
    #pragma unroll
    for (int j = 0; j < 4; ++j)
      acc[j] = __builtin_amdgcn_wmma_f32_16x16x32_bf16(false, a0.v, false, b0[j].v,
                                                       (short)0, acc[j], false, false);
    if (k0 + 64 < K) {                           // prefetch stage 0 of next iter
      loadA(a0, k0 + 64);
      loadB(b0, k0 + 64);
    }
    #pragma unroll
    for (int j = 0; j < 4; ++j)
      acc[j] = __builtin_amdgcn_wmma_f32_16x16x32_bf16(false, a1.v, false, b1[j].v,
                                                       (short)0, acc[j], false, false);
  }

  #pragma unroll
  for (int j = 0; j < 4; ++j) {
    int col = n0 + 16 * j + l15;
    float bv = bias ? bias[col] : 0.f;
    #pragma unroll
    for (int r = 0; r < 8; ++r) {
      int row = m0 + half * 8 + r;
      float v = acc[j][r] + bv;
      if (relu) v = fmaxf(v, 0.f);
      Cf[(size_t)row * Nout + col] = v;
    }
  }
}

// ---------- small utility kernels ----------
__global__ void k_f32_to_bf16(const float* __restrict__ in, bf16* __restrict__ out, size_t n) {
  size_t i = (size_t)blockIdx.x * blockDim.x + threadIdx.x;
  if (i < n) out[i] = (bf16)in[i];
}
// W [K][Nout] f32 -> WT [Nout][K] bf16
__global__ void k_transpose_bf16(const float* __restrict__ W, bf16* __restrict__ WT, int K, int Nout) {
  int i = blockIdx.x * blockDim.x + threadIdx.x;
  if (i < K * Nout) {
    int k = i / Nout, n = i - k * Nout;
    WT[(size_t)n * K + k] = (bf16)W[i];
  }
}
__global__ void k_degree(const int* __restrict__ dst, float* __restrict__ deg, int E) {
  int e = blockIdx.x * blockDim.x + threadIdx.x;
  if (e < E) atomicAdd(&deg[dst[e]], 1.0f);
}
// scatter-add of projected features, float4 per thread
__global__ void k_scatter_add4(const int* __restrict__ src, const int* __restrict__ dst,
                               const float* __restrict__ Y, float* __restrict__ agg,
                               int D4, size_t total) {
  size_t i = (size_t)blockIdx.x * blockDim.x + threadIdx.x;
  if (i < total) {
    size_t e = i / (size_t)D4;
    int d = (int)(i - e * (size_t)D4) * 4;
    int D = D4 * 4;
    const float4 v = *(const float4*)(Y + (size_t)src[e] * D + d);
    float* ap = agg + (size_t)dst[e] * D + d;
    atomicAdd(ap + 0, v.x);
    atomicAdd(ap + 1, v.y);
    atomicAdd(ap + 2, v.z);
    atomicAdd(ap + 3, v.w);
  }
}
// out = relu(agg/deg + p) -> bf16
__global__ void k_combine(const float* __restrict__ agg, const float* __restrict__ p,
                          const float* __restrict__ deg, bf16* __restrict__ out,
                          int D, size_t total) {
  size_t i = (size_t)blockIdx.x * blockDim.x + threadIdx.x;
  if (i < total) {
    size_t n = i / (size_t)D;
    float v = agg[i] / fmaxf(deg[n], 1.0f) + p[i];
    out[i] = (bf16)fmaxf(v, 0.f);
  }
}
// a_s[n][h] = sum_c hg[n][h][c]*att_src[h][c]; same for a_d
__global__ void k_att_scores(const float* __restrict__ hg, const float* __restrict__ asrc,
                             const float* __restrict__ adst, float* __restrict__ a_s,
                             float* __restrict__ a_d) {
  int i = blockIdx.x * blockDim.x + threadIdx.x;
  if (i < NN * NH) {
    int n = i / NH, h = i - n * NH;
    const float* p = hg + (size_t)n * DG + h * DD3;
    float ss = 0.f, sd = 0.f;
    for (int c = 0; c < DD3; ++c) { ss += p[c] * asrc[h * DD3 + c]; sd += p[c] * adst[h * DD3 + c]; }
    a_s[i] = ss; a_d[i] = sd;
  }
}
// thread per edge: 4 heads
__global__ void k_edge_max(const int* __restrict__ src, const int* __restrict__ dst,
                           const float* __restrict__ a_s, const float* __restrict__ a_d,
                           unsigned* __restrict__ m, int E) {
  int e = blockIdx.x * blockDim.x + threadIdx.x;
  if (e < E) {
    int s = src[e], d = dst[e];
    #pragma unroll
    for (int h = 0; h < NH; ++h) {
      float v = lrelu(a_s[s * NH + h] + a_d[d * NH + h]);
      atomicMax(&m[d * NH + h], f2ord(v));
    }
  }
}
__global__ void k_self_max(const float* __restrict__ a_s, const float* __restrict__ a_d,
                           unsigned* __restrict__ m) {
  int i = blockIdx.x * blockDim.x + threadIdx.x;
  if (i < NN * NH) atomicMax(&m[i], f2ord(lrelu(a_s[i] + a_d[i])));
}
__global__ void k_decode_max(unsigned* __restrict__ m) {
  int i = blockIdx.x * blockDim.x + threadIdx.x;
  if (i < NN * NH) {
    float v = ord2f(m[i]);
    ((float*)m)[i] = v;
  }
}
// block per item (edge or self loop); 128 threads: h = t>>5, c = (t&31)*4
__global__ void k_gat_acc(const int* __restrict__ src, const int* __restrict__ dst,
                          const float* __restrict__ hg, const float* __restrict__ a_s,
                          const float* __restrict__ a_d, const float* __restrict__ mf,
                          float* __restrict__ denom, float* __restrict__ numer, int self_mode) {
  int e = blockIdx.x;
  int s, d;
  if (self_mode) { s = e; d = e; } else { s = src[e]; d = dst[e]; }
  int t = threadIdx.x;
  int h = t >> 5, c = (t & 31) * 4;
  float ex = __expf(lrelu(a_s[s * NH + h] + a_d[d * NH + h]) - mf[d * NH + h]);
  if ((t & 31) == 0) atomicAdd(&denom[d * NH + h], ex);
  const float4 v = *(const float4*)(hg + (size_t)s * DG + h * DD3 + c);
  float* np = numer + (size_t)d * DG + h * DD3 + c;
  atomicAdd(np + 0, ex * v.x);
  atomicAdd(np + 1, ex * v.y);
  atomicAdd(np + 2, ex * v.z);
  atomicAdd(np + 3, ex * v.w);
}
__global__ void k_gat_final(const float* __restrict__ numer, const float* __restrict__ denom,
                            const float* __restrict__ bg, float* __restrict__ zout,
                            bf16* __restrict__ zbf) {
  int i = blockIdx.x * blockDim.x + threadIdx.x;
  if (i < NN * DD3) {
    int n = i / DD3, c = i - n * DD3;
    float s = 0.f;
    #pragma unroll
    for (int h = 0; h < NH; ++h)
      s += numer[(size_t)n * DG + h * DD3 + c] / denom[n * NH + h];
    s = s * 0.25f + bg[c];
    zout[i] = s;
    zbf[i] = (bf16)s;
  }
}
// out[n][o] = sum_j t[n][j]*W2[j][o] + b2[o]   (64 -> Dout, tiny)
__global__ void k_head2(const float* __restrict__ t, const float* __restrict__ W2,
                        const float* __restrict__ b2, float* __restrict__ out, int Dout) {
  int i = blockIdx.x * blockDim.x + threadIdx.x;
  if (i < NN * Dout) {
    int n = i / Dout, o = i - n * Dout;
    float s = b2[o];
    const float* tr = t + (size_t)n * 64;
    for (int j = 0; j < 64; ++j) s += tr[j] * W2[j * Dout + o];
    out[i] = s;
  }
}

extern "C" void kernel_launch(void* const* d_in, const int* in_sizes, int n_in,
                              void* d_out, int out_size, void* d_ws, size_t ws_size,
                              hipStream_t stream) {
  (void)in_sizes; (void)n_in; (void)out_size; (void)ws_size;
  const float* x   = (const float*)d_in[0];
  const int*   ei  = (const int*)d_in[1];
  const float* W1l = (const float*)d_in[2];
  const float* b1l = (const float*)d_in[3];
  const float* W1r = (const float*)d_in[4];
  const float* W2l = (const float*)d_in[5];
  const float* b2l = (const float*)d_in[6];
  const float* W2r = (const float*)d_in[7];
  const float* W3l = (const float*)d_in[8];
  const float* b3l = (const float*)d_in[9];
  const float* W3r = (const float*)d_in[10];
  const float* Wg  = (const float*)d_in[11];
  const float* bg  = (const float*)d_in[12];
  const float* att_src = (const float*)d_in[13];
  const float* att_dst = (const float*)d_in[14];
  const float* Wd1 = (const float*)d_in[15];
  const float* bd1 = (const float*)d_in[16];
  const float* Wd2 = (const float*)d_in[17];
  const float* bd2 = (const float*)d_in[18];
  const float* Wq1 = (const float*)d_in[19];
  const float* bq1 = (const float*)d_in[20];
  const float* Wq2 = (const float*)d_in[21];
  const float* bq2 = (const float*)d_in[22];
  const float* Ws1 = (const float*)d_in[23];
  const float* bs1 = (const float*)d_in[24];
  const float* Ws2 = (const float*)d_in[25];
  const float* bs2 = (const float*)d_in[26];
  const int* srcI = ei;
  const int* dstI = ei + EE;

  float* out_delta = (float*)d_out;            // [NN]
  float* out_qual  = out_delta + NN;           // [NN]
  float* out_strat = out_delta + 2 * NN;       // [NN*4]
  float* out_emb   = out_delta + 6 * NN;       // [NN*128]

  char* base = (char*)d_ws;
  size_t off = 0;
  auto alloc = [&](size_t bytes) -> void* {
    void* p = base + off;
    off = (off + bytes + 255) & ~(size_t)255;
    return p;
  };
  // bf16 transposed weights
  bf16* W1lT = (bf16*)alloc((size_t)HID * DIN * 2);
  bf16* W1rT = (bf16*)alloc((size_t)HID * DIN * 2);
  bf16* W2lT = (bf16*)alloc((size_t)HID * HID * 2);
  bf16* W2rT = (bf16*)alloc((size_t)HID * HID * 2);
  bf16* W3lT = (bf16*)alloc((size_t)DD3 * HID * 2);
  bf16* W3rT = (bf16*)alloc((size_t)DD3 * HID * 2);
  bf16* WgT  = (bf16*)alloc((size_t)DG * DD3 * 2);
  bf16* Wd1T = (bf16*)alloc((size_t)64 * DD3 * 2);
  bf16* Wq1T = (bf16*)alloc((size_t)64 * DD3 * 2);
  bf16* Ws1T = (bf16*)alloc((size_t)64 * DD3 * 2);
  // activations / scratch
  float* deg    = (float*)alloc((size_t)NN * 4);
  bf16*  xbf    = (bf16*)alloc((size_t)NN * DIN * 2);
  float* bufY   = (float*)alloc((size_t)NN * DG * 4);   // y projections; later hg [NN][512]
  float* bufP   = (float*)alloc((size_t)NN * HID * 4);  // self-path projections
  float* bufAgg = (float*)alloc((size_t)NN * HID * 4);  // scatter accumulators
  bf16*  h1bf   = (bf16*)alloc((size_t)NN * HID * 2);
  bf16*  h2bf   = (bf16*)alloc((size_t)NN * HID * 2);
  bf16*  h3bf   = (bf16*)alloc((size_t)NN * DD3 * 2);
  float* numer  = (float*)alloc((size_t)NN * DG * 4);
  float* a_s    = (float*)alloc((size_t)NN * NH * 4);
  float* a_d    = (float*)alloc((size_t)NN * NH * 4);
  unsigned* mB  = (unsigned*)alloc((size_t)NN * NH * 4);
  float* denom  = (float*)alloc((size_t)NN * NH * 4);
  bf16*  zbf    = (bf16*)alloc((size_t)NN * DD3 * 2);
  float* t      = (float*)alloc((size_t)NN * 64 * 4);
  float* hg = bufY;          // reuse after SAGE layers
  float* mf = (float*)mB;

  auto gemm = [&](const bf16* A, const bf16* BT, const float* bias, int relu,
                  float* Cf, int M, int Nout, int K) {
    int tiles = (M / 16) * (Nout / 64);
    gemm_bf16_wmma<<<cdiv_h(tiles, 8), 256, 0, stream>>>(A, BT, bias, relu, Cf, M, Nout, K);
  };
  auto sage = [&](const bf16* hin, const bf16* WlT, const bf16* WrT, const float* bl,
                  bf16* hout, int Din, int Dout) {
    gemm(hin, WlT, nullptr, 0, bufY, NN, Dout, Din);          // neighbor projection
    gemm(hin, WrT, bl, 0, bufP, NN, Dout, Din);               // self path + bias
    hipMemsetAsync(bufAgg, 0, (size_t)NN * Dout * 4, stream);
    k_scatter_add4<<<cdiv_h((size_t)EE * (Dout / 4), 256), 256, 0, stream>>>(
        srcI, dstI, bufY, bufAgg, Dout / 4, (size_t)EE * (Dout / 4));
    k_combine<<<cdiv_h((size_t)NN * Dout, 256), 256, 0, stream>>>(
        bufAgg, bufP, deg, hout, Dout, (size_t)NN * Dout);
  };

  // ---- weight prep ----
  k_transpose_bf16<<<cdiv_h((size_t)DIN * HID, 256), 256, 0, stream>>>(W1l, W1lT, DIN, HID);
  k_transpose_bf16<<<cdiv_h((size_t)DIN * HID, 256), 256, 0, stream>>>(W1r, W1rT, DIN, HID);
  k_transpose_bf16<<<cdiv_h((size_t)HID * HID, 256), 256, 0, stream>>>(W2l, W2lT, HID, HID);
  k_transpose_bf16<<<cdiv_h((size_t)HID * HID, 256), 256, 0, stream>>>(W2r, W2rT, HID, HID);
  k_transpose_bf16<<<cdiv_h((size_t)HID * DD3, 256), 256, 0, stream>>>(W3l, W3lT, HID, DD3);
  k_transpose_bf16<<<cdiv_h((size_t)HID * DD3, 256), 256, 0, stream>>>(W3r, W3rT, HID, DD3);
  k_transpose_bf16<<<cdiv_h((size_t)DD3 * DG, 256), 256, 0, stream>>>(Wg, WgT, DD3, DG);
  k_transpose_bf16<<<cdiv_h((size_t)DD3 * 64, 256), 256, 0, stream>>>(Wd1, Wd1T, DD3, 64);
  k_transpose_bf16<<<cdiv_h((size_t)DD3 * 64, 256), 256, 0, stream>>>(Wq1, Wq1T, DD3, 64);
  k_transpose_bf16<<<cdiv_h((size_t)DD3 * 64, 256), 256, 0, stream>>>(Ws1, Ws1T, DD3, 64);

  // ---- degree + input conversion ----
  hipMemsetAsync(deg, 0, (size_t)NN * 4, stream);
  k_degree<<<cdiv_h(EE, 256), 256, 0, stream>>>(dstI, deg, EE);
  k_f32_to_bf16<<<cdiv_h((size_t)NN * DIN, 256), 256, 0, stream>>>(x, xbf, (size_t)NN * DIN);

  // ---- SAGE stack (project-then-aggregate) ----
  sage(xbf,  W1lT, W1rT, b1l, h1bf, DIN, HID);   // 768 -> 256
  sage(h1bf, W2lT, W2rT, b2l, h2bf, HID, HID);   // 256 -> 256
  sage(h2bf, W3lT, W3rT, b3l, h3bf, HID, DD3);   // 256 -> 128

  // ---- GAT: 128 -> 4x128 projection, segment softmax, head mean ----
  gemm(h3bf, WgT, nullptr, 0, hg, NN, DG, DD3);
  k_att_scores<<<cdiv_h(NN * NH, 256), 256, 0, stream>>>(hg, att_src, att_dst, a_s, a_d);
  hipMemsetAsync(mB, 0, (size_t)NN * NH * 4, stream);        // 0 == ordered(-inf)
  k_edge_max<<<cdiv_h(EE, 256), 256, 0, stream>>>(srcI, dstI, a_s, a_d, mB, EE);
  k_self_max<<<cdiv_h(NN * NH, 256), 256, 0, stream>>>(a_s, a_d, mB);
  k_decode_max<<<cdiv_h(NN * NH, 256), 256, 0, stream>>>(mB);
  hipMemsetAsync(denom, 0, (size_t)NN * NH * 4, stream);
  hipMemsetAsync(numer, 0, (size_t)NN * DG * 4, stream);
  k_gat_acc<<<EE, 128, 0, stream>>>(srcI, dstI, hg, a_s, a_d, mf, denom, numer, 0);
  k_gat_acc<<<NN, 128, 0, stream>>>(srcI, dstI, hg, a_s, a_d, mf, denom, numer, 1);
  k_gat_final<<<cdiv_h((size_t)NN * DD3, 256), 256, 0, stream>>>(numer, denom, bg, out_emb, zbf);

  // ---- heads ----
  gemm(zbf, Wd1T, bd1, 1, t, NN, 64, DD3);
  k_head2<<<cdiv_h(NN * 1, 256), 256, 0, stream>>>(t, Wd2, bd2, out_delta, 1);
  gemm(zbf, Wq1T, bq1, 1, t, NN, 64, DD3);
  k_head2<<<cdiv_h(NN * 1, 256), 256, 0, stream>>>(t, Wq2, bq2, out_qual, 1);
  gemm(zbf, Ws1T, bs1, 1, t, NN, 64, DD3);
  k_head2<<<cdiv_h(NN * NH, 256), 256, 0, stream>>>(t, Ws2, bs2, out_strat, NH);
}
